// CollaQMultiHeadAttention_31009663877567
// MI455X (gfx1250) — matrix operand order repaired and checked by reading, not compiled, not run on hardware
//
#include <hip/hip_runtime.h>
#include <hip/hip_bf16.h>
#include <stdint.h>

// ---------------------------------------------------------------------------
// CDNA5 (gfx1250) fused MHA block: proj+LN -> flash attention -> FFN
// All GEMMs use v_wmma_f32_16x16x32_f16 (f16 in, f32 accumulate).
// ---------------------------------------------------------------------------

typedef __attribute__((ext_vector_type(16))) _Float16 v16h;
typedef __attribute__((ext_vector_type(8)))  _Float16 v8h;
typedef __attribute__((ext_vector_type(8)))  float    v8f;

#define WMMA16(a, b, c) \
  __builtin_amdgcn_wmma_f32_16x16x32_f16(false, (a), false, (b), (short)0, (c), false, false)

// Contiguous 16-half load (two 16B LDS reads) -> B fragment.
__device__ __forceinline__ v16h ld16h(const _Float16* p) {
  union { v16h v; v8h h[2]; } u;
  u.h[0] = *(const v8h*)(p);
  u.h[1] = *(const v8h*)(p + 8);
  return u.v;
}

// A fragment (16x32 f16): lanes<16 hold K {0..15},
// lanes>=16 hold K {16..31} of row (lane%16). `row` points at a
// 32-half contiguous row; hi = lane>>4.
__device__ __forceinline__ v16h ldAfrag(const _Float16* row, int hi) {
  union { v16h v; v8h h[2]; } u;
  u.h[0] = *(const v8h*)(row + hi * 16);
  u.h[1] = *(const v8h*)(row + hi * 16 + 8);
  return u.v;
}

// Pack two f32 -> packed-f16 dword.
__device__ __forceinline__ unsigned pk2(float a, float b) {
  union { _Float16 h[2]; unsigned w; } u;
  u.h[0] = (_Float16)a;
  u.h[1] = (_Float16)b;
  return u.w;
}

// ---------------------------------------------------------------------------
// Kernel 1: X[8192,512] @ W[512,512] + bias, then LayerNorm(512) -> f16
// normalized output in [B,H,L,64] layout; optional fp32 residual (pre-norm).
// Block: 256 threads (8 waves), 32 output rows x 512 cols. Wave w owns cols
// [64w, 64w+64) x all 32 rows (acc[2][4]). LayerNorm stats are computed from
// the accumulators via 16-lane shuffles + an 8-wave LDS combine (no f32
// LDS staging round-trip).
// ---------------------------------------------------------------------------
__global__ __launch_bounds__(256) void proj_ln_kernel(
    const float* __restrict__ X, const float* __restrict__ W,
    const float* __restrict__ bias, const float* __restrict__ gamma,
    const float* __restrict__ beta, _Float16* __restrict__ outN,
    float* __restrict__ resid) {
  const int rowbase = blockIdx.x * 32;
  const int lane = threadIdx.x & 31;
  const int wv   = threadIdx.x >> 5;
  const int lm   = lane & 15;
  const int hi   = lane >> 4;

  __shared__ __align__(32) _Float16 Asub[32 * 32];   // 2 KB
  __shared__ __align__(32) _Float16 Bt[512 * 32];    // 32 KB: Bt[n*32+k]
  __shared__ float red_s[8][32], red_ss[8][32];
  __shared__ float mus[32], rns[32];

  v8f acc[2][4];
#pragma unroll
  for (int jr = 0; jr < 2; ++jr)
#pragma unroll
    for (int jc = 0; jc < 4; ++jc)
#pragma unroll
      for (int r = 0; r < 8; ++r) acc[jr][jc][r] = 0.f;

  for (int ks = 0; ks < 16; ++ks) {
    __syncthreads();
    // Stage A tile 32x32 (f32 -> f16, packed 8B DS stores).
    {
      int e  = threadIdx.x * 4;          // 1024 elements
      int ar = e >> 5, ak = e & 31;
      float4 a4 = *(const float4*)&X[(size_t)(rowbase + ar) * 512 + ks * 32 + ak];
      uint2 p;
      p.x = pk2(a4.x, a4.y);
      p.y = pk2(a4.z, a4.w);
      *(uint2*)&Asub[ar * 32 + ak] = p;
    }
    // Stage W tile 32x512 transposed as packed k-pairs: Bt[n*32+k] = W[k][n].
#pragma unroll
    for (int i = 0; i < 8; ++i) {
      int f  = threadIdx.x + 256 * i;    // 2048 pair-jobs
      int k2 = (f >> 7) * 2;             // even k in 0..30
      int bc = (f & 127) * 4;            // n in 0..508
      float4 wa = *(const float4*)&W[(size_t)(ks * 32 + k2) * 512 + bc];
      float4 wb = *(const float4*)&W[(size_t)(ks * 32 + k2 + 1) * 512 + bc];
      *(unsigned*)&Bt[(bc + 0) * 32 + k2] = pk2(wa.x, wb.x);
      *(unsigned*)&Bt[(bc + 1) * 32 + k2] = pk2(wa.y, wb.y);
      *(unsigned*)&Bt[(bc + 2) * 32 + k2] = pk2(wa.z, wb.z);
      *(unsigned*)&Bt[(bc + 3) * 32 + k2] = pk2(wa.w, wb.w);
    }
    __syncthreads();
    v16h a0 = ldAfrag(&Asub[lm * 32], hi);
    v16h a1 = ldAfrag(&Asub[(16 + lm) * 32], hi);
#pragma unroll
    for (int jc = 0; jc < 4; ++jc) {
      v16h bf = ld16h(&Bt[(wv * 64 + jc * 16 + lm) * 32 + hi * 16]);
      acc[0][jc] = WMMA16(a0, bf, acc[0][jc]);
      acc[1][jc] = WMMA16(a1, bf, acc[1][jc]);
    }
  }

  // Bias add (per-lane columns).
  float bcol[4], gcol[4], becol[4];
#pragma unroll
  for (int jc = 0; jc < 4; ++jc) {
    int col = wv * 64 + jc * 16 + lm;
    bcol[jc]  = bias[col];
    gcol[jc]  = gamma[col];
    becol[jc] = beta[col];
  }
#pragma unroll
  for (int jr = 0; jr < 2; ++jr)
#pragma unroll
    for (int jc = 0; jc < 4; ++jc)
#pragma unroll
      for (int r = 0; r < 8; ++r) acc[jr][jc][r] += bcol[jc];

  // Residual store (uniform branch, pre-norm values).
  if (resid != nullptr) {
#pragma unroll
    for (int jr = 0; jr < 2; ++jr)
#pragma unroll
      for (int jc = 0; jc < 4; ++jc) {
        int col = wv * 64 + jc * 16 + lm;
#pragma unroll
        for (int r = 0; r < 8; ++r) {
          int row = rowbase + 16 * jr + r + 8 * hi;
          resid[(size_t)row * 512 + col] = acc[jr][jc][r];
        }
      }
  }

  // LayerNorm stats: per-row sum/sumsq over this wave's 64 cols, then
  // 16-lane xor-reduce (C rows live on half-waves), then 8-wave combine.
#pragma unroll
  for (int jr = 0; jr < 2; ++jr) {
#pragma unroll
    for (int r = 0; r < 8; ++r) {
      float s = 0.f, ss = 0.f;
#pragma unroll
      for (int jc = 0; jc < 4; ++jc) {
        float v = acc[jr][jc][r];
        s += v; ss += v * v;
      }
#pragma unroll
      for (int off = 8; off; off >>= 1) {
        s  += __shfl_xor(s, off);
        ss += __shfl_xor(ss, off);
      }
      if (lm == 0) {
        int rl = 16 * jr + r + 8 * hi;
        red_s[wv][rl]  = s;
        red_ss[wv][rl] = ss;
      }
    }
  }
  __syncthreads();
  if (threadIdx.x < 32) {
    int rl = threadIdx.x;
    float ts = 0.f, tss = 0.f;
#pragma unroll
    for (int w = 0; w < 8; ++w) { ts += red_s[w][rl]; tss += red_ss[w][rl]; }
    float mu  = ts * (1.0f / 512.0f);
    float var = tss * (1.0f / 512.0f) - mu * mu;
    mus[rl] = mu;
    rns[rl] = rsqrtf(var + 1e-6f);
  }
  __syncthreads();

  // Normalize from registers and store f16 in [B,H,L,64] layout.
#pragma unroll
  for (int jr = 0; jr < 2; ++jr) {
#pragma unroll
    for (int r = 0; r < 8; ++r) {
      int rl   = 16 * jr + r + 8 * hi;
      float mu = mus[rl], rn = rns[rl];
      int grow = rowbase + rl;
      int bb = grow >> 10, ll = grow & 1023;
#pragma unroll
      for (int jc = 0; jc < 4; ++jc) {
        int c = wv * 64 + jc * 16 + lm;
        float y = (acc[jr][jc][r] - mu) * rn * gcol[jc] + becol[jc];
        int hh = c >> 6, dd = c & 63;
        outN[(((size_t)(bb * 8 + hh) * 1024) + ll) * 64 + dd] = (_Float16)y;
      }
    }
  }
}

// ---------------------------------------------------------------------------
// Kernel 2: flash attention. Grid (L/128, B*H). Block 256 = 8 waves; wave w
// owns q-rows [128*bx + 16w, +16). Online softmax over k-tiles of 32.
// ---------------------------------------------------------------------------
__global__ __launch_bounds__(256) void attn_kernel(
    const _Float16* __restrict__ Qn, const _Float16* __restrict__ Kn,
    const _Float16* __restrict__ Vn, _Float16* __restrict__ O) {
  const int bh = blockIdx.y;           // 0..63
  const int b = bh >> 3, h = bh & 7;
  const int qbase = blockIdx.x * 128;
  const int lane = threadIdx.x & 31;
  const int wv   = threadIdx.x >> 5;
  const int lm   = lane & 15;
  const int hi   = lane >> 4;

  __shared__ __align__(32) _Float16 Qs[128 * 64];     // 16 KB
  __shared__ __align__(32) _Float16 Ks[32 * 64];      //  4 KB
  __shared__ __align__(32) _Float16 Vt[64 * 32];      //  4 KB (transposed V)
  __shared__ __align__(32) _Float16 Ps[8 * 16 * 32];  //  8 KB P scratch

  const _Float16* qptr = Qn + (size_t)bh * 1024 * 64;
  const _Float16* kptr = Kn + (size_t)bh * 1024 * 64;
  const _Float16* vptr = Vn + (size_t)bh * 1024 * 64;

  // Stage Q tile 128x64.
#pragma unroll
  for (int i = 0; i < 4; ++i) {
    int U = threadIdx.x + 256 * i;     // uint4 index, 1024 total
    int r = U >> 3, c8 = U & 7;
    *(uint4*)&Qs[r * 64 + c8 * 8] = *(const uint4*)&qptr[(size_t)(qbase + r) * 64 + c8 * 8];
  }
  __syncthreads();
  v16h aq0 = ldAfrag(&Qs[(wv * 16 + lm) * 64 + 0], hi);
  v16h aq1 = ldAfrag(&Qs[(wv * 16 + lm) * 64 + 32], hi);

  float mrow[8], lrow[8];
  v8f acc[4];
#pragma unroll
  for (int r = 0; r < 8; ++r) { mrow[r] = -1e30f; lrow[r] = 0.f; }
#pragma unroll
  for (int j = 0; j < 4; ++j)
#pragma unroll
    for (int r = 0; r < 8; ++r) acc[j][r] = 0.f;

  for (int kt = 0; kt < 32; ++kt) {
    __syncthreads();
    // Split staging: threads 0-127 build transposed V tile (packed k-pair
    // b32 DS stores), threads 128-255 copy the K tile (b128).
    if (threadIdx.x < 128) {
      int rp = threadIdx.x >> 3;       // k-row pair 0..15
      int c8 = threadIdx.x & 7;
      uint4 va = *(const uint4*)&vptr[(size_t)(kt * 32 + 2 * rp) * 64 + c8 * 8];
      uint4 vb = *(const uint4*)&vptr[(size_t)(kt * 32 + 2 * rp + 1) * 64 + c8 * 8];
      const _Float16* ha = (const _Float16*)&va;
      const _Float16* hb = (const _Float16*)&vb;
#pragma unroll
      for (int u = 0; u < 8; ++u) {
        union { _Float16 h[2]; unsigned w; } pk;
        pk.h[0] = ha[u];
        pk.h[1] = hb[u];
        *(unsigned*)&Vt[(c8 * 8 + u) * 32 + 2 * rp] = pk.w;
      }
    } else {
      int t2 = threadIdx.x - 128;
#pragma unroll
      for (int i = 0; i < 2; ++i) {
        int U = t2 + 128 * i;          // 256 uint4
        int r = U >> 3, c8 = U & 7;
        *(uint4*)&Ks[r * 64 + c8 * 8] =
            *(const uint4*)&kptr[(size_t)(kt * 32 + r) * 64 + c8 * 8];
      }
    }
    __syncthreads();

    // S = Q . K^T over this 32-key tile: two 16x16 C fragments.
    v8f s0, s1;
#pragma unroll
    for (int r = 0; r < 8; ++r) { s0[r] = 0.f; s1[r] = 0.f; }
    {
      v16h bk = ld16h(&Ks[(0 * 16 + lm) * 64 + 0 + 16 * hi]);
      s0 = WMMA16(aq0, bk, s0);
      bk = ld16h(&Ks[(0 * 16 + lm) * 64 + 32 + 16 * hi]);
      s0 = WMMA16(aq1, bk, s0);
      bk = ld16h(&Ks[(1 * 16 + lm) * 64 + 0 + 16 * hi]);
      s1 = WMMA16(aq0, bk, s1);
      bk = ld16h(&Ks[(1 * 16 + lm) * 64 + 32 + 16 * hi]);
      s1 = WMMA16(aq1, bk, s1);
    }

    _Float16* pp = &Ps[wv * 512];
#pragma unroll
    for (int r = 0; r < 8; ++r) {
      float x0 = s0[r] * 0.125f;       // 1/sqrt(DK)
      float x1 = s1[r] * 0.125f;
      float mt = fmaxf(x0, x1);
#pragma unroll
      for (int off = 8; off; off >>= 1) mt = fmaxf(mt, __shfl_xor(mt, off));
      float mn   = fmaxf(mrow[r], mt);
      float corr = __expf(mrow[r] - mn);
      mrow[r] = mn;
      float p0 = __expf(x0 - mn);
      float p1 = __expf(x1 - mn);
      float psum = p0 + p1;
#pragma unroll
      for (int off = 8; off; off >>= 1) psum += __shfl_xor(psum, off);
      lrow[r] = lrow[r] * corr + psum;
#pragma unroll
      for (int j = 0; j < 4; ++j) acc[j][r] *= corr;
      pp[(r + 8 * hi) * 32 + lm]      = (_Float16)p0;
      pp[(r + 8 * hi) * 32 + 16 + lm] = (_Float16)p1;
    }
    // Wave-local LDS RAW fence before re-reading P as an A fragment.
    asm volatile("s_wait_dscnt 0" ::: "memory");
    v16h ap = ldAfrag(&pp[lm * 32], hi);
#pragma unroll
    for (int j = 0; j < 4; ++j) {
      v16h bv = ld16h(&Vt[(j * 16 + lm) * 32 + 16 * hi]);
      acc[j] = WMMA16(ap, bv, acc[j]);
    }
  }

  // Normalize and store O (f16) in [B*L, H*64] layout.
#pragma unroll
  for (int j = 0; j < 4; ++j) {
#pragma unroll
    for (int r = 0; r < 8; ++r) {
      int row = qbase + wv * 16 + r + 8 * hi;
      float ov = acc[j][r] / lrow[r];
      O[(size_t)(b * 1024 + row) * 512 + h * 64 + j * 16 + lm] = (_Float16)ov;
    }
  }
}

// ---------------------------------------------------------------------------
// Kernel 3: generic A(f16)[8192,512] @ W(f32)[512,512] + bias, optional ReLU.
// Block: 32 rows x 512 cols, 8 waves. Writes f16 (outH) or f32 (outF).
// ---------------------------------------------------------------------------
__global__ __launch_bounds__(256) void gemm_kernel(
    const _Float16* __restrict__ A, const float* __restrict__ W,
    const float* __restrict__ bias, float* __restrict__ outF,
    _Float16* __restrict__ outH, int relu) {
  const int rowbase = blockIdx.x * 32;
  const int lane = threadIdx.x & 31;
  const int wv   = threadIdx.x >> 5;
  const int lm   = lane & 15;
  const int hi   = lane >> 4;

  __shared__ __align__(32) _Float16 Asub[32 * 32];   // 2 KB
  __shared__ __align__(32) _Float16 Bt[512 * 32];    // 32 KB

  v8f acc[2][4];
#pragma unroll
  for (int jr = 0; jr < 2; ++jr)
#pragma unroll
    for (int jc = 0; jc < 4; ++jc)
#pragma unroll
      for (int r = 0; r < 8; ++r) acc[jr][jc][r] = 0.f;

  for (int ks = 0; ks < 16; ++ks) {
    __syncthreads();
    // Stage A tile 32x32 (f16 copy, 8B DS stores).
    {
      int e  = threadIdx.x * 4;
      int ar = e >> 5, ak = e & 31;
      *(uint2*)&Asub[ar * 32 + ak] =
          *(const uint2*)&A[(size_t)(rowbase + ar) * 512 + ks * 32 + ak];
    }
    // Stage W tile transposed, packed k-pairs.
#pragma unroll
    for (int i = 0; i < 8; ++i) {
      int f  = threadIdx.x + 256 * i;
      int k2 = (f >> 7) * 2;
      int bc = (f & 127) * 4;
      float4 wa = *(const float4*)&W[(size_t)(ks * 32 + k2) * 512 + bc];
      float4 wb = *(const float4*)&W[(size_t)(ks * 32 + k2 + 1) * 512 + bc];
      *(unsigned*)&Bt[(bc + 0) * 32 + k2] = pk2(wa.x, wb.x);
      *(unsigned*)&Bt[(bc + 1) * 32 + k2] = pk2(wa.y, wb.y);
      *(unsigned*)&Bt[(bc + 2) * 32 + k2] = pk2(wa.z, wb.z);
      *(unsigned*)&Bt[(bc + 3) * 32 + k2] = pk2(wa.w, wb.w);
    }
    __syncthreads();
    v16h a0 = ldAfrag(&Asub[lm * 32], hi);
    v16h a1 = ldAfrag(&Asub[(16 + lm) * 32], hi);
#pragma unroll
    for (int jc = 0; jc < 4; ++jc) {
      v16h bf = ld16h(&Bt[(wv * 64 + jc * 16 + lm) * 32 + hi * 16]);
      acc[0][jc] = WMMA16(a0, bf, acc[0][jc]);
      acc[1][jc] = WMMA16(a1, bf, acc[1][jc]);
    }
  }

  if (outF != nullptr) {
#pragma unroll
    for (int jr = 0; jr < 2; ++jr)
#pragma unroll
      for (int jc = 0; jc < 4; ++jc) {
        int col = wv * 64 + jc * 16 + lm;
        float bv = bias[col];
#pragma unroll
        for (int r = 0; r < 8; ++r) {
          int row = rowbase + 16 * jr + r + 8 * hi;
          float v = acc[jr][jc][r] + bv;
          if (relu) v = fmaxf(v, 0.f);
          outF[(size_t)row * 512 + col] = v;
        }
      }
  } else {
#pragma unroll
    for (int jr = 0; jr < 2; ++jr)
#pragma unroll
      for (int jc = 0; jc < 4; ++jc) {
        int col = wv * 64 + jc * 16 + lm;
        float bv = bias[col];
#pragma unroll
        for (int r = 0; r < 8; ++r) {
          int row = rowbase + 16 * jr + r + 8 * hi;
          float v = acc[jr][jc][r] + bv;
          if (relu) v = fmaxf(v, 0.f);
          outH[(size_t)row * 512 + col] = (_Float16)v;
        }
      }
  }
}

// ---------------------------------------------------------------------------
extern "C" void kernel_launch(void* const* d_in, const int* in_sizes, int n_in,
                              void* d_out, int out_size, void* d_ws, size_t ws_size,
                              hipStream_t stream) {
  (void)in_sizes; (void)n_in; (void)out_size; (void)ws_size;
  const float* q    = (const float*)d_in[0];
  const float* k    = (const float*)d_in[1];
  const float* v    = (const float*)d_in[2];
  const float* Wq   = (const float*)d_in[3];
  const float* bq   = (const float*)d_in[4];
  const float* Wk   = (const float*)d_in[5];
  const float* bk   = (const float*)d_in[6];
  const float* Wv   = (const float*)d_in[7];
  const float* bv   = (const float*)d_in[8];
  const float* g_q  = (const float*)d_in[9];
  const float* be_q = (const float*)d_in[10];
  const float* g_k  = (const float*)d_in[11];
  const float* be_k = (const float*)d_in[12];
  const float* g_v  = (const float*)d_in[13];
  const float* be_v = (const float*)d_in[14];
  const float* W1   = (const float*)d_in[15];
  const float* b1   = (const float*)d_in[16];
  const float* W2   = (const float*)d_in[17];
  const float* b2   = (const float*)d_in[18];

  float* outp  = (float*)d_out;                      // [8,1024,512]
  float* resid = outp + (size_t)8 * 1024 * 512;      // residual [8,1024,8,64]

  const size_t HB = (size_t)8 * 8 * 1024 * 64 * sizeof(_Float16);  // 8 MB
  char* ws = (char*)d_ws;
  _Float16* qn   = (_Float16*)(ws + 0 * HB);   // [B,H,L,64]
  _Float16* kn   = (_Float16*)(ws + 1 * HB);
  _Float16* vn   = (_Float16*)(ws + 2 * HB);
  _Float16* attn = (_Float16*)(ws + 3 * HB);   // [B*L, 512]
  _Float16* hbuf = (_Float16*)(ws + 4 * HB);   // [B*L, 512]

  dim3 blk(256);
  proj_ln_kernel<<<dim3(256), blk, 0, stream>>>(q, Wq, bq, g_q, be_q, qn, resid);
  proj_ln_kernel<<<dim3(256), blk, 0, stream>>>(k, Wk, bk, g_k, be_k, kn, nullptr);
  proj_ln_kernel<<<dim3(256), blk, 0, stream>>>(v, Wv, bv, g_v, be_v, vn, nullptr);
  attn_kernel<<<dim3(8, 64), blk, 0, stream>>>(qn, kn, vn, attn);
  gemm_kernel<<<dim3(256), blk, 0, stream>>>(attn, W1, b1, nullptr, hbuf, 1);
  gemm_kernel<<<dim3(256), blk, 0, stream>>>(hbuf, W2, b2, outp, nullptr, 0);
}
